// SACRF_26594437496903
// MI455X (gfx1250) — matrix operand order repaired and checked by reading, not compiled, not run on hardware
//
#include <hip/hip_runtime.h>
#include <hip/hip_bf16.h>

// ---------------- problem constants ----------------
#define BB   64
#define TT   10
#define NN   12
#define DD   1024
#define HH   8
#define DHD  128
#define AA   9
#define MTOK (BB*TT*NN)            // 7680 tokens
#define MDSZ ((size_t)MTOK*DD)     // 7,864,320 elements

#define KC   128                   // K-chunk staged per TDM transfer
#define BROW 136                   // padded LDS row: 128 f16 data + 8 f16 pad (272B)

typedef __attribute__((ext_vector_type(16))) _Float16 v16h;
typedef __attribute__((ext_vector_type(8)))  _Float16 v8h;
typedef __attribute__((ext_vector_type(8)))  float    v8f;
typedef __attribute__((ext_vector_type(4)))  unsigned v4u;
typedef __attribute__((ext_vector_type(8)))  int      v8i32;
typedef __attribute__((ext_vector_type(4)))  int      v4i32;

#if defined(__AMDGCN__) && defined(__has_builtin)
#if __has_builtin(__builtin_amdgcn_tensor_load_to_lds) && __has_builtin(__builtin_amdgcn_s_wait_tensorcnt)
#define HAVE_TDM 1
#endif
#endif
#ifndef HAVE_TDM
#define HAVE_TDM 0
#endif

#if HAVE_TDM
// Issue one TDM 2-D tile load: Bt[n0..n0+63][k0..k0+KC-1] (f16) -> LDS (padded rows).
// D# per CDNA5 ISA ch.8: group0 = {count/flags, lds_addr, global_addr lo, hi|type},
// group1 = {datasize/pad, dims, tile dims, strides}.  Uniform (SGPR) values.
__device__ __forceinline__ void tdm_issue_b(const _Float16* gptr, unsigned lds_off,
                                            int K, int NR)
{
  unsigned long long ga = (unsigned long long)gptr;
  v4u g0;
  g0[0] = 1u;                                            // count=1, user mode
  g0[1] = lds_off;                                       // LDS dest byte address
  g0[2] = (unsigned)ga;                                  // global_addr[31:0]
  g0[3] = (unsigned)((ga >> 32) & 0x1FFFFFFu) | (2u << 30); // addr[56:32] | type=2
  v8i32 g1;
  g1[0] = (int)((1u << 16)      // data_size = 2 bytes
              | (1u << 20)      // pad_enable
              | (5u << 22)      // pad_interval: 64 DWORDs (one 256B row)
              | (3u << 25));    // pad_amount: 4 DWORDs (16B) -> 272B row stride
  g1[1] = (int)(((unsigned)K & 0xFFFFu) << 16);                    // tensor_dim0 lo
  g1[2] = (int)(((unsigned)K >> 16) | (((unsigned)NR & 0xFFFFu) << 16)); // dim0 hi|dim1 lo
  g1[3] = (int)(((unsigned)NR >> 16) | ((unsigned)KC << 16));      // dim1 hi | tile_dim0
  g1[4] = 64;                                                      // tile_dim1 (rows)
  g1[5] = (int)K;                                                  // dim0_stride[31:0]
  g1[6] = (int)(((unsigned)K & 0xFFFFu) << 16);                    // stride0 hi16=0 | stride1 lo16
  g1[7] = (int)((unsigned)K >> 16);                                // stride1 hi
  v4i32 z4 = {0, 0, 0, 0};
#if __clang_major__ >= 23
  v8i32 z8 = {0, 0, 0, 0, 0, 0, 0, 0};
  __builtin_amdgcn_tensor_load_to_lds(g0, g1, z4, z4, z8, 0);
#else
  __builtin_amdgcn_tensor_load_to_lds(g0, g1, z4, z4, 0);
#endif
}
#endif

// ---------------------------------------------------------------------------
// WMMA GEMM:  out[M,Nc] = A[M,K](f16) * Bt[Nc,K]^T(f16)   (Bt = B transposed)
// Block = 8 waves -> 128x64 output tile. B panel (64 cols x 128 k) staged in
// LDS by TDM (double buffered, overlapped with compute); each wave computes a
// 16x64 tile with 4 v_wmma_f32_16x16x32_f16 per 32-k step, B frags from LDS.
// op: 0=f16 store | 1=+bias,relu,f16 | 2=+bias,f32 | 3=+bias,S+=w*val | 4=f32
// ---------------------------------------------------------------------------
__global__ void __launch_bounds__(256) k_gemm(
    const _Float16* __restrict__ A, const _Float16* __restrict__ Bt,
    const float* __restrict__ bias, float* __restrict__ outF,
    _Float16* __restrict__ outH, float* __restrict__ accum,
    const float* __restrict__ wvec, int widx,
    int M, int K, int Nc, int op)
{
  __shared__ _Float16 bs[2][64 * BROW];

  const int tilesN = Nc >> 6;
  const int bm   = (blockIdx.x / tilesN) << 7;   // 128-row block base
  const int n0   = (blockIdx.x % tilesN) << 6;   // 64-col base
  const int wave = threadIdx.x >> 5;
  const int lane = threadIdx.x & 31;
  const int half = lane >> 4;
  const int r    = lane & 15;
  const int m0   = bm + wave * 16;

  // A frag rows: lane half 0 holds K[0..15]; half 1 K[16..31]
  const _Float16* ap = A + (size_t)(m0 + r) * K + half * 16;
  const int nchunk = K / KC;

#if HAVE_TDM
  if (wave == 0) {
    tdm_issue_b(Bt + (size_t)n0 * K, (unsigned)(size_t)(void*)&bs[0][0], K, Nc);
    __builtin_amdgcn_s_wait_tensorcnt(0);
  }
  __syncthreads();
#endif

  v8f c0 = {}, c1 = {}, c2 = {}, c3 = {};
  for (int c = 0; c < nchunk; ++c) {
    const int kbase = c * KC;
#if HAVE_TDM
    if (wave == 0 && c + 1 < nchunk)
      tdm_issue_b(Bt + (size_t)n0 * K + (kbase + KC),
                  (unsigned)(size_t)(void*)&bs[(c + 1) & 1][0], K, Nc);
#else
    { // cooperative copy fallback: 256 threads move 64 rows x 128 f16
      const int row = threadIdx.x >> 2, seg = (threadIdx.x & 3) * 32;
      const _Float16* src = Bt + (size_t)(n0 + row) * K + kbase + seg;
      _Float16* dst = &bs[c & 1][row * BROW + seg];
      *(v8h*)(dst +  0) = *(const v8h*)(src +  0);
      *(v8h*)(dst +  8) = *(const v8h*)(src +  8);
      *(v8h*)(dst + 16) = *(const v8h*)(src + 16);
      *(v8h*)(dst + 24) = *(const v8h*)(src + 24);
      __syncthreads();
    }
#endif
    __builtin_prefetch(ap + kbase + KC, 0, 1);         // global_prefetch_b8 (speculative)

    // load the chunk's 4 A fragments up-front (overlap with LDS-fed WMMAs)
    v16h a[4];
    #pragma unroll
    for (int s = 0; s < 4; ++s) {
      const _Float16* p = ap + kbase + s * 32;
      v8h lo = *(const v8h*)p;
      v8h hi = *(const v8h*)(p + 8);
      a[s] = __builtin_shufflevector(lo, hi, 0,1,2,3,4,5,6,7,8,9,10,11,12,13,14,15);
    }

    const _Float16* ls = &bs[c & 1][0];
    #pragma unroll
    for (int s = 0; s < 4; ++s) {
      const int ko = s * 32 + half * 16;               // f16 offset within padded row
      v8h l0 = *(const v8h*)(ls + (r +  0) * BROW + ko);
      v8h h0 = *(const v8h*)(ls + (r +  0) * BROW + ko + 8);
      v8h l1 = *(const v8h*)(ls + (r + 16) * BROW + ko);
      v8h h1 = *(const v8h*)(ls + (r + 16) * BROW + ko + 8);
      v8h l2 = *(const v8h*)(ls + (r + 32) * BROW + ko);
      v8h h2 = *(const v8h*)(ls + (r + 32) * BROW + ko + 8);
      v8h l3 = *(const v8h*)(ls + (r + 48) * BROW + ko);
      v8h h3 = *(const v8h*)(ls + (r + 48) * BROW + ko + 8);
      v16h b0 = __builtin_shufflevector(l0, h0, 0,1,2,3,4,5,6,7,8,9,10,11,12,13,14,15);
      v16h b1 = __builtin_shufflevector(l1, h1, 0,1,2,3,4,5,6,7,8,9,10,11,12,13,14,15);
      v16h b2 = __builtin_shufflevector(l2, h2, 0,1,2,3,4,5,6,7,8,9,10,11,12,13,14,15);
      v16h b3 = __builtin_shufflevector(l3, h3, 0,1,2,3,4,5,6,7,8,9,10,11,12,13,14,15);
      c0 = __builtin_amdgcn_wmma_f32_16x16x32_f16(false, a[s], false, b0, (short)0, c0, false, false);
      c1 = __builtin_amdgcn_wmma_f32_16x16x32_f16(false, a[s], false, b1, (short)0, c1, false, false);
      c2 = __builtin_amdgcn_wmma_f32_16x16x32_f16(false, a[s], false, b2, (short)0, c2, false, false);
      c3 = __builtin_amdgcn_wmma_f32_16x16x32_f16(false, a[s], false, b3, (short)0, c3, false, false);
    }

#if HAVE_TDM
    if (wave == 0 && c + 1 < nchunk) __builtin_amdgcn_s_wait_tensorcnt(0);
    __syncthreads();
#else
    __syncthreads();
#endif
  }

  // D layout: VGPR v -> row m0 + half*8 + v ; col = n0 + t*16 + r
  const int   rowb = m0 + half * 8;
  const float wv   = (op == 3) ? wvec[widx] : 0.0f;
  v8f accs[4] = {c0, c1, c2, c3};
  #pragma unroll
  for (int t = 0; t < 4; ++t) {
    const int col  = n0 + t * 16 + r;
    const float bv = (op == 1 || op == 2 || op == 3) ? bias[col] : 0.0f;
    #pragma unroll
    for (int v = 0; v < 8; ++v) {
      float  val = accs[t][v];
      size_t idx = (size_t)(rowb + v) * Nc + col;
      if      (op == 0) { outH[idx] = (_Float16)val; }
      else if (op == 1) { val += bv; val = val > 0.f ? val : 0.f; outH[idx] = (_Float16)val; }
      else if (op == 2) { outF[idx] = val + bv; }
      else if (op == 3) { accum[idx] += wv * (val + bv); }
      else              { outF[idx] = val; }
    }
  }
}

// --------------- pos embedding + residual add, write f32 & f16 ---------------
__global__ void __launch_bounds__(256) k_posadd(
    const float* __restrict__ feat, const float* __restrict__ boxes,
    float* __restrict__ F, _Float16* __restrict__ Fh)
{
  const size_t t = blockIdx.x;
  const float cx = (boxes[t*4+0] + boxes[t*4+2]) * 0.5f * (1280.0f/157.0f);
  const float cy = (boxes[t*4+1] + boxes[t*4+3]) * 0.5f * (720.0f/87.0f);
  for (int d = threadIdx.x; d < DD; d += 256) {
    const float c   = (d < 512) ? cx : cy;
    const int   dd  = d & 511;
    const float ex  = (2.0f * (float)(dd >> 1)) / 512.0f;
    const float ph  = c * __powf(10000.0f, -ex);
    const float pe  = (dd & 1) ? __cosf(ph) : __sinf(ph);
    const float v   = feat[t*DD + d] + pe;
    F[t*DD + d]  = v;
    Fh[t*DD + d] = (_Float16)v;
  }
}

// ---------- softmax-free multihead attention core (c <= 12, DH = 128) --------
__global__ void __launch_bounds__(128) k_attn(
    const _Float16* __restrict__ TX, const _Float16* __restrict__ PX,
    const _Float16* __restrict__ GX, _Float16* __restrict__ O,
    int c, int temporal)
{
  __shared__ float st[12*DHD], sp[12*DHD], sg[12*DHD], sa[144];
  const int g = blockIdx.x, h = blockIdx.y, e = threadIdx.x;
  for (int l = 0; l < c; ++l) {
    int t;
    if (temporal) { int b = g / NN, n = g - b*NN; t = (b*TT + l)*NN + n; }
    else          { t = g*c + l; }
    const size_t base = (size_t)t*DD + h*DHD + e;
    st[l*DHD+e] = (float)TX[base];
    sp[l*DHD+e] = (float)PX[base];
    sg[l*DHD+e] = (float)GX[base];
  }
  __syncthreads();
  for (int i = e; i < c*c; i += 128) {
    const int l = i / c, m = i - l*c;
    float s = 0.f;
    for (int q = 0; q < DHD; ++q) s += st[l*DHD+q] * sp[m*DHD+q];
    sa[i] = s * 0.03125f;                       // 1/sqrt(1024)
  }
  __syncthreads();
  for (int l = 0; l < c; ++l) {
    float s = 0.f;
    for (int m = 0; m < c; ++m) s += sa[l*c+m] * sg[m*DHD+e];
    int t;
    if (temporal) { int b = g / NN, n = g - b*NN; t = (b*TT + l)*NN + n; }
    else          { t = g*c + l; }
    O[(size_t)t*DD + h*DHD + e] = (_Float16)s;
  }
}

// --------------- layernorm( P + F ) * gamma + beta -> f16 --------------------
__global__ void __launch_bounds__(256) k_ln(
    const float* __restrict__ P, const float* __restrict__ F,
    const float* __restrict__ g, const float* __restrict__ b,
    _Float16* __restrict__ Y)
{
  __shared__ float rs[256], rq[256];
  const size_t t = blockIdx.x;
  float s = 0.f, q = 0.f;
  for (int d = threadIdx.x; d < DD; d += 256) {
    const float v = P[t*DD+d] + F[t*DD+d];
    s += v; q += v*v;
  }
  rs[threadIdx.x] = s; rq[threadIdx.x] = q;
  __syncthreads();
  for (int k = 128; k > 0; k >>= 1) {
    if (threadIdx.x < k) { rs[threadIdx.x] += rs[threadIdx.x+k]; rq[threadIdx.x] += rq[threadIdx.x+k]; }
    __syncthreads();
  }
  const float mean = rs[0] * (1.0f/DD);
  const float var  = rq[0] * (1.0f/DD) - mean*mean;
  const float inv  = rsqrtf(var + 1e-5f);
  for (int d = threadIdx.x; d < DD; d += 256) {
    const float v = P[t*DD+d] + F[t*DD+d];
    Y[t*DD+d] = (_Float16)(((v - mean) * inv) * g[d] + b[d]);
  }
}

// -------- small projections to A=9 actions: out[t,a] = X[t,:]·W[:,a]+b -------
__global__ void k_proj9(const float* __restrict__ X, const float* __restrict__ W,
                        const float* __restrict__ bias, float* __restrict__ out)
{
  const int i = blockIdx.x * blockDim.x + threadIdx.x;
  if (i >= MTOK * AA) return;
  const int t = i / AA, a = i - t*AA;
  const float* x = X + (size_t)t * DD;
  float s = bias ? bias[a] : 0.f;
  for (int d = 0; d < DD; ++d) s += x[d] * W[d*AA + a];
  out[i] = s;
}

// ------- Q_hat = softmax(Q_u + Qsp@ct_s + Qtp@ct_t) over A=9 -----------------
__global__ void k_qhat(const float* __restrict__ Qu, const float* __restrict__ Qsp,
                       const float* __restrict__ Qtp, const float* __restrict__ cts,
                       const float* __restrict__ ctt, float* __restrict__ out)
{
  const int t = blockIdx.x * blockDim.x + threadIdx.x;
  if (t >= MTOK) return;
  float q[AA];
  for (int a = 0; a < AA; ++a) {
    float s = Qu[t*AA+a];
    for (int j = 0; j < AA; ++j)
      s += Qsp[t*AA+j]*cts[j*AA+a] + Qtp[t*AA+j]*ctt[j*AA+a];
    q[a] = s;
  }
  float mx = q[0];
  for (int a = 1; a < AA; ++a) mx = q[a] > mx ? q[a] : mx;
  float se = 0.f;
  for (int a = 0; a < AA; ++a) { q[a] = __expf(q[a]-mx); se += q[a]; }
  const float is = 1.0f / se;
  for (int a = 0; a < AA; ++a) out[t*AA+a] = q[a]*is;
}

// ---- C_v = S + Tm (to d_out); per-token halt prob p, halt step k; Kmax ------
__global__ void __launch_bounds__(256) k_st(
    const float* __restrict__ S, const float* __restrict__ Tm,
    const float* __restrict__ hW, const float* __restrict__ hB,
    float* __restrict__ cv, float* __restrict__ p_arr,
    int* __restrict__ k_arr, int* __restrict__ kmax)
{
  __shared__ float red[256];
  const size_t t = blockIdx.x;
  float dot = 0.f;
  for (int d = threadIdx.x; d < DD; d += 256) {
    const float v = S[t*DD+d] + Tm[t*DD+d];
    cv[t*DD+d] = v;
    dot += v * hW[d];
  }
  red[threadIdx.x] = dot; __syncthreads();
  for (int k = 128; k > 0; k >>= 1) {
    if (threadIdx.x < k) red[threadIdx.x] += red[threadIdx.x+k];
    __syncthreads();
  }
  if (threadIdx.x == 0) {
    const float p = 1.f / (1.f + __expf(-(red[0] + hB[0])));
    float hp = 0.f; int kk = 11;                 // 11 == never halts in 10 steps
    for (int j = 1; j <= 10; ++j) { const float hn = hp + p; if (hn >= 1.f) { kk = j; break; } hp = hn; }
    p_arr[t] = p; k_arr[t] = kk;
    atomicMax(kmax, kk < 10 ? kk : 10);          // J = min(10, max k)
  }
}

// -------- halt loss: R_t + sum(N_t), with run-gated step count J -------------
__global__ void __launch_bounds__(256) k_halt(
    const float* __restrict__ p_arr, const int* __restrict__ k_arr,
    const int* __restrict__ kmax, float* __restrict__ loss)
{
  __shared__ float red[256];
  const int t = blockIdx.x * 256 + threadIdx.x;
  float local = 0.f;
  if (t < MTOK) {
    const int   J = *kmax;
    const float p = p_arr[t];
    const int   k = k_arr[t];
    int nadd, rc;
    if (k <= J) { nadd = k - 1; rc = J - k + 1; }
    else        { nadd = J;     rc = 0; }
    float hp = 0.f;
    for (int j = 0; j < nadd; ++j) hp += p;      // match fp32 iterative accumulation
    const int Nt = (rc == 0) ? 10 : (11 - rc);
    local = (1.f - hp) + (float)Nt;
  }
  red[threadIdx.x] = local; __syncthreads();
  for (int k = 128; k > 0; k >>= 1) {
    if (threadIdx.x < k) red[threadIdx.x] += red[threadIdx.x+k];
    __syncthreads();
  }
  if (threadIdx.x == 0) atomicAdd(loss, red[0]);
}

// -------- weight converters fp32 -> f16 transposed [N,K] ---------------------
__global__ void k_convT(const float* __restrict__ W, _Float16* __restrict__ Bt,
                        int K, int Nc)
{
  const int i = blockIdx.x * blockDim.x + threadIdx.x;
  if (i >= K * Nc) return;
  const int n = i / K, k = i - n * K;
  Bt[i] = (_Float16)W[(size_t)k * Nc + n];
}
__global__ void k_convHeads(const float* __restrict__ Wh, _Float16* __restrict__ Bt)
{
  // Wh: [H,D,DH]; effective W[d][h*128+e] = Wh[h][d][e]; store transposed [n][k]
  const int i = blockIdx.x * blockDim.x + threadIdx.x;
  if (i >= DD * DD) return;
  const int n = i / DD, k = i - n * DD;
  const int h = n >> 7, e = n & 127;
  Bt[i] = (_Float16)Wh[(size_t)h * DD * DHD + (size_t)k * DHD + e];
}

// -------- zero S accumulator, loss slot, Kmax --------------------------------
__global__ void k_zero(float* __restrict__ S, float* __restrict__ loss,
                       int* __restrict__ kmax)
{
  const size_t i = (size_t)blockIdx.x * blockDim.x + threadIdx.x;
  if (i < MDSZ) S[i] = 0.f;
  if (i == 0) { *loss = 0.f; *kmax = 0; }
}

// ============================================================================
extern "C" void kernel_launch(void* const* d_in, const int* in_sizes, int n_in,
                              void* d_out, int out_size, void* d_ws, size_t ws_size,
                              hipStream_t stream)
{
  struct Blk { const float *theta,*phi,*g,*Wout,*ln_g,*ln_b,*f1W,*f1b,*f2W,*f2b; };
  const float *features,*boxes,*f_u_W,*f_u_b,*f_sp_W,*f_tp_W,*ct_s,*ct_t,*halt_W,*halt_b,*sp_w;
  Blk sp, tp;
  auto In = [&](int i){ return (const float*)d_in[i]; };

  if (in_sizes[0] == 30720) {
    // jax pytree (alphabetically sorted) flattening
    boxes=In(0); features=In(1); ct_s=In(2); ct_t=In(3); f_sp_W=In(4); f_tp_W=In(5);
    f_u_W=In(6); f_u_b=In(7); halt_W=In(8); halt_b=In(9);
    sp = {In(19),In(18),In(15),In(10),In(17),In(16),In(11),In(12),In(13),In(14)};
    sp_w = In(20);
    tp = {In(30),In(29),In(26),In(21),In(28),In(27),In(22),In(23),In(24),In(25)};
  } else {
    // dict insertion order
    features=In(0); boxes=In(1); f_u_W=In(2); f_u_b=In(3);
    sp = {In(4),In(5),In(6),In(7),In(8),In(9),In(10),In(11),In(12),In(13)};
    sp_w = In(14);
    tp = {In(15),In(16),In(17),In(18),In(19),In(20),In(21),In(22),In(23),In(24)};
    f_sp_W=In(25); f_tp_W=In(26); ct_s=In(27); ct_t=In(28); halt_W=In(29); halt_b=In(30);
  }

  // ---------------- workspace layout ----------------
  char* base = (char*)d_ws; size_t off = 0;
  auto alloc = [&](size_t bytes)->void* {
    void* p = base + off; off = (off + bytes + 255) & ~(size_t)255; return p;
  };
  float*    F    = (float*)   alloc(MDSZ*4);
  _Float16* Fh   = (_Float16*)alloc(MDSZ*2);
  _Float16* wt[12];                               // sp:{th,ph,g,wo,f1,f2} tp:{...}
  for (int i = 0; i < 12; ++i) wt[i] = (_Float16*)alloc((size_t)DD*DD*2);
  _Float16* TXh  = (_Float16*)alloc(MDSZ*2);
  _Float16* PXh  = (_Float16*)alloc(MDSZ*2);
  _Float16* GXh  = (_Float16*)alloc(MDSZ*2);
  _Float16* Oh   = (_Float16*)alloc(MDSZ*2);
  _Float16* Yh   = (_Float16*)alloc(MDSZ*2);
  _Float16* Rh   = (_Float16*)alloc(MDSZ*2);
  float*    P    = (float*)   alloc(MDSZ*4);
  float*    S    = (float*)   alloc(MDSZ*4);
  float*    Tm   = (float*)   alloc(MDSZ*4);
  float*    Qu   = (float*)   alloc((size_t)MTOK*AA*4);
  float*    Qsp  = (float*)   alloc((size_t)MTOK*AA*4);
  float*    Qtp  = (float*)   alloc((size_t)MTOK*AA*4);
  float*    p_arr= (float*)   alloc((size_t)MTOK*4);
  int*      kArr = (int*)     alloc((size_t)MTOK*4);
  int*      kMax = (int*)     alloc(256);

  float* out      = (float*)d_out;
  float* out_qhat = out;
  float* out_cv   = out + (size_t)MTOK*AA;
  float* out_loss = out + (size_t)MTOK*AA + MDSZ;

  const int cBlk  = (MTOK/128) * (DD/64);           // 60*16 = 960 GEMM blocks
  const int cConv = (DD*DD + 255) / 256;

  auto gemm = [&](const _Float16* A, const _Float16* Bt, const float* bias,
                  float* oF, _Float16* oH, float* acc, const float* wv, int wi, int op) {
    k_gemm<<<cBlk, 256, 0, stream>>>(A, Bt, bias, oF, oH, acc, wv, wi, MTOK, DD, DD, op);
  };

  // ---------------- pipeline ----------------
  k_zero<<<(int)((MDSZ + 255)/256), 256, 0, stream>>>(S, out_loss, kMax);
  k_posadd<<<MTOK, 256, 0, stream>>>(features, boxes, F, Fh);

  // weight conversion (theta/phi/g via head-packing, rest plain transpose)
  k_convHeads<<<cConv, 256, 0, stream>>>(sp.theta, wt[0]);
  k_convHeads<<<cConv, 256, 0, stream>>>(sp.phi,   wt[1]);
  k_convHeads<<<cConv, 256, 0, stream>>>(sp.g,     wt[2]);
  k_convT   <<<cConv, 256, 0, stream>>>(sp.Wout, wt[3], DD, DD);
  k_convT   <<<cConv, 256, 0, stream>>>(sp.f1W,  wt[4], DD, DD);
  k_convT   <<<cConv, 256, 0, stream>>>(sp.f2W,  wt[5], DD, DD);
  k_convHeads<<<cConv, 256, 0, stream>>>(tp.theta, wt[6]);
  k_convHeads<<<cConv, 256, 0, stream>>>(tp.phi,   wt[7]);
  k_convHeads<<<cConv, 256, 0, stream>>>(tp.g,     wt[8]);
  k_convT   <<<cConv, 256, 0, stream>>>(tp.Wout, wt[9],  DD, DD);
  k_convT   <<<cConv, 256, 0, stream>>>(tp.f1W,  wt[10], DD, DD);
  k_convT   <<<cConv, 256, 0, stream>>>(tp.f2W,  wt[11], DD, DD);

  k_proj9<<<(MTOK*AA + 255)/256, 256, 0, stream>>>(F, f_u_W, f_u_b, Qu);

  // ---- spatial: shared projections, then per-clique attention + block_post
  gemm(Fh, wt[0], nullptr, nullptr, TXh, nullptr, nullptr, 0, 0);
  gemm(Fh, wt[1], nullptr, nullptr, PXh, nullptr, nullptr, 0, 0);
  gemm(Fh, wt[2], nullptr, nullptr, GXh, nullptr, nullptr, 0, 0);
  const int cliques[4] = {2, 3, 6, 12};
  for (int e = 0; e < 4; ++e) {
    const int c = cliques[e];
    k_attn<<<dim3(MTOK/c, HH), 128, 0, stream>>>(TXh, PXh, GXh, Oh, c, 0);
    gemm(Oh, wt[3], nullptr, P, nullptr, nullptr, nullptr, 0, 4);
    k_ln<<<MTOK, 256, 0, stream>>>(P, F, sp.ln_g, sp.ln_b, Yh);
    gemm(Yh, wt[4], sp.f1b, nullptr, Rh, nullptr, nullptr, 0, 1);
    gemm(Rh, wt[5], sp.f2b, nullptr, nullptr, S, sp_w, e, 3);   // S += w[e]*(..)
  }

  // ---- temporal
  gemm(Fh, wt[6], nullptr, nullptr, TXh, nullptr, nullptr, 0, 0);
  gemm(Fh, wt[7], nullptr, nullptr, PXh, nullptr, nullptr, 0, 0);
  gemm(Fh, wt[8], nullptr, nullptr, GXh, nullptr, nullptr, 0, 0);
  k_attn<<<dim3(BB*NN, HH), 128, 0, stream>>>(TXh, PXh, GXh, Oh, TT, 1);
  gemm(Oh, wt[9], nullptr, P, nullptr, nullptr, nullptr, 0, 4);
  k_ln<<<MTOK, 256, 0, stream>>>(P, F, tp.ln_g, tp.ln_b, Yh);
  gemm(Yh, wt[10], tp.f1b, nullptr, Rh, nullptr, nullptr, 0, 1);
  gemm(Rh, wt[11], tp.f2b, Tm, nullptr, nullptr, nullptr, 0, 2);

  // ---- Q heads + softmax
  k_proj9<<<(MTOK*AA + 255)/256, 256, 0, stream>>>(S,  f_sp_W, nullptr, Qsp);
  k_proj9<<<(MTOK*AA + 255)/256, 256, 0, stream>>>(Tm, f_tp_W, nullptr, Qtp);
  k_qhat <<<(MTOK + 255)/256, 256, 0, stream>>>(Qu, Qsp, Qtp, ct_s, ct_t, out_qhat);

  // ---- C_v = S+Tm, halting closed form
  k_st  <<<MTOK, 256, 0, stream>>>(S, Tm, halt_W, halt_b, out_cv, p_arr, kArr, kMax);
  k_halt<<<(MTOK + 255)/256, 256, 0, stream>>>(p_arr, kArr, kMax, out_loss);
}
